// PointNet_Estimation_1726576857089
// MI455X (gfx1250) — matrix operand order repaired and checked by reading, not compiled, not run on hardware
//
#include <hip/hip_runtime.h>
#include <cstddef>

typedef __attribute__((ext_vector_type(16))) _Float16 v16h;
typedef __attribute__((ext_vector_type(8)))  _Float16 v8h;
typedef __attribute__((ext_vector_type(8)))  float    v8f;

// ---------------------------------------------------------------- utilities
__global__ void zero_kernel(float* p, int n) {
  int i = blockIdx.x * blockDim.x + threadIdx.x;
  if (i < n) p[i] = 0.f;
}

// (B,3,N) -> (B,N,3)
__global__ void transpose_xyz_kernel(const float* __restrict__ in,
                                     float* __restrict__ out, int B, int N) {
  int i = blockIdx.x * blockDim.x + threadIdx.x;
  if (i >= B * N) return;
  int b = i / N, n = i - b * N;
  const float* src = in + (size_t)b * 3 * N + n;
  float* dst = out + (size_t)i * 3;
  dst[0] = src[0];
  dst[1] = src[(size_t)N];
  dst[2] = src[(size_t)2 * N];
}

// ------------------------------------------------------- farthest point sample
__global__ void fps_kernel(const float* __restrict__ pts, int N, int npoint,
                           int* __restrict__ outIdx) {
  __shared__ float dist[4096];
  __shared__ float rv[256];
  __shared__ int   ri[256];
  int b = blockIdx.x;
  int tid = threadIdx.x;
  const float* p = pts + (size_t)b * N * 3;
  for (int n = tid; n < N; n += blockDim.x) dist[n] = 1e10f;
  __syncthreads();
  int far = 0;
  for (int it = 0; it < npoint; ++it) {
    if (tid == 0) outIdx[b * npoint + it] = far;
    float cx = p[far * 3 + 0], cy = p[far * 3 + 1], cz = p[far * 3 + 2];
    float bv = -1.f;
    int bi = 0;
    for (int n = tid; n < N; n += blockDim.x) {
      float dx = p[n * 3 + 0] - cx;
      float dy = p[n * 3 + 1] - cy;
      float dz = p[n * 3 + 2] - cz;
      float d = dx * dx + dy * dy + dz * dz;
      float dn = fminf(dist[n], d);
      dist[n] = dn;
      if (dn > bv) { bv = dn; bi = n; }
    }
    rv[tid] = bv; ri[tid] = bi;
    __syncthreads();
    for (int o = blockDim.x >> 1; o > 0; o >>= 1) {
      if (tid < o) {
        if (rv[tid + o] > rv[tid] ||
            (rv[tid + o] == rv[tid] && ri[tid + o] < ri[tid])) {
          rv[tid] = rv[tid + o]; ri[tid] = ri[tid + o];
        }
      }
      __syncthreads();
    }
    far = ri[0];
    __syncthreads();
  }
}

__global__ void gather_xyz_kernel(const float* __restrict__ pts,
                                  const int* __restrict__ idx,
                                  float* __restrict__ out, int N, int M, int B) {
  int i = blockIdx.x * blockDim.x + threadIdx.x;
  if (i >= B * M) return;
  int b = i / M;
  int n = idx[i];
  const float* s = pts + ((size_t)b * N + n) * 3;
  float* d = out + (size_t)i * 3;
  d[0] = s[0]; d[1] = s[1]; d[2] = s[2];
}

// ------------------------------------------------------------- ball query
__global__ void ballquery_kernel(const float* __restrict__ pts,
                                 const float* __restrict__ centers,
                                 int* __restrict__ idx,
                                 int B, int N, int M, int K, float r2) {
  int i = blockIdx.x * blockDim.x + threadIdx.x;
  if (i >= B * M) return;
  int b = i / M;
  const float* p = pts + (size_t)b * N * 3;
  float cx = centers[(size_t)i * 3 + 0];
  float cy = centers[(size_t)i * 3 + 1];
  float cz = centers[(size_t)i * 3 + 2];
  int* row = idx + (size_t)i * K;
  int cnt = 0;
  for (int n = 0; n < N && cnt < K; ++n) {
    float dx = p[n * 3 + 0] - cx;
    float dy = p[n * 3 + 1] - cy;
    float dz = p[n * 3 + 2] - cz;
    float d = dx * dx + dy * dy + dz * dz;
    if (d <= r2) row[cnt++] = n;
  }
  int f = (cnt > 0) ? row[0] : 0;
  for (; cnt < K; ++cnt) row[cnt] = f;
}

// grouped row = (b*M+m)*K + k ; channels = [feat(Cfeat), xyz-rel(3), 0-pad]
__global__ void group_kernel(const float* __restrict__ pts,
                             const _Float16* __restrict__ feat,
                             const float* __restrict__ centers,
                             const int* __restrict__ idx,
                             _Float16* __restrict__ out,
                             int B, int N, int M, int K, int Cfeat, int Cpad) {
  int i = blockIdx.x * blockDim.x + threadIdx.x;
  int total = B * M * K;
  if (i >= total) return;
  int bm = i / K;
  int b = bm / M;
  int n = idx[i];
  _Float16* o = out + (size_t)i * Cpad;
  const float* pp = pts + ((size_t)b * N + n) * 3;
  const float* cc = centers + (size_t)bm * 3;
  if (Cfeat > 0) {
    const _Float16* ff = feat + ((size_t)b * N + n) * Cfeat;
    for (int c = 0; c < Cfeat; ++c) o[c] = ff[c];
  }
  o[Cfeat + 0] = (_Float16)(pp[0] - cc[0]);
  o[Cfeat + 1] = (_Float16)(pp[1] - cc[1]);
  o[Cfeat + 2] = (_Float16)(pp[2] - cc[2]);
  for (int c = Cfeat + 3; c < Cpad; ++c) o[c] = (_Float16)0.f;
}

// SA3 input: [xyz(3), feat(640), 0-pad] per (b, point), stride 672
__global__ void concat_sa3_kernel(const float* __restrict__ nx2,
                                  const _Float16* __restrict__ l2,
                                  _Float16* __restrict__ out, int total) {
  int i = blockIdx.x * blockDim.x + threadIdx.x;
  if (i >= total) return;
  _Float16* o = out + (size_t)i * 672;
  o[0] = (_Float16)nx2[(size_t)i * 3 + 0];
  o[1] = (_Float16)nx2[(size_t)i * 3 + 1];
  o[2] = (_Float16)nx2[(size_t)i * 3 + 2];
  const _Float16* f = l2 + (size_t)i * 640;
  for (int c = 0; c < 640; ++c) o[3 + c] = f[c];
  for (int c = 643; c < 672; ++c) o[c] = (_Float16)0.f;
}

// --------------------------------------------------------------- W packing
// P layout: [tileN][kchunk][lane(32)][h(16)] halves, in the contiguous k order the GEMM also uses for A, a B-matrix
// 32x16 f16 fragment: col = tileN*16 + (lane&15); k = kc*32 + (lane>>4)*16 + h.
__global__ void pack_w_kernel(const float* __restrict__ W,
                              _Float16* __restrict__ P,
                              int K, int N, int Kpad) {
  int numKc = Kpad >> 5;
  int total = (N >> 4) * numKc * 512;
  int i = blockIdx.x * blockDim.x + threadIdx.x;
  if (i >= total) return;
  int h = i & 15;
  int lane = (i >> 4) & 31;
  int kc = (i >> 9) % numKc;
  int tn = (i >> 9) / numKc;
  int col = (tn << 4) + (lane & 15);
  int k = (kc << 5) + ((lane >> 4) << 4) + h;
  P[i] = (k < K) ? (_Float16)W[(size_t)k * N + col] : (_Float16)0.f;
}

// ------------------------------------------------------------ WMMA GEMM+bias
// C[R,N](f16) = A[R,Kpad](f16) @ Wpack + bias(f32).
// R mult of 16; Kpad mult of 32; N mult of 16. No guards in the k-loop:
// 4 aligned b128 loads + 1 v_wmma_f32_16x16x32_f16 per 32-k chunk.
__global__ void wmma_gemm_bias_kernel(const _Float16* __restrict__ A,
                                      const _Float16* __restrict__ P,
                                      const float* __restrict__ bias,
                                      _Float16* __restrict__ C,
                                      int R, int Kpad, int N) {
  int lane = threadIdx.x & 31;
  int wave = threadIdx.x >> 5;
  int tileRow = blockIdx.x * 16;
  int tileCol = (blockIdx.y * 4 + wave) * 16;
  if (tileCol >= N) return;  // wave-uniform exit; EXEC stays all-1 for WMMA
  int numKc = Kpad >> 5;
  int rowA = tileRow + (lane & 15);
  int hi = lane >> 4;  // 0 or 1
  const v8h* Ar = (const v8h*)(A + (size_t)rowA * Kpad) + hi * 2;
  const v8h* Bp = (const v8h*)(P + (size_t)(tileCol >> 4) * numKc * 512) + lane * 2;
  v8f acc = {};
  for (int kc = 0; kc < numKc; ++kc) {
    v8h a0 = Ar[kc * 4 + 0];
    v8h a1 = Ar[kc * 4 + 1];
    v8h b0 = Bp[kc * 64 + 0];
    v8h b1 = Bp[kc * 64 + 1];
    v16h a = __builtin_shufflevector(a0, a1, 0, 1, 2, 3, 4, 5, 6, 7,
                                     8, 9, 10, 11, 12, 13, 14, 15);
    v16h b = __builtin_shufflevector(b0, b1, 0, 1, 2, 3, 4, 5, 6, 7,
                                     8, 9, 10, 11, 12, 13, 14, 15);
    acc = __builtin_amdgcn_wmma_f32_16x16x32_f16(
        false, a, false, b, (short)0, acc, false, false);
  }
  int colB = tileCol + (lane & 15);
  float bv = bias[colB];
  int rbase = tileRow + (hi << 3);
#pragma unroll
  for (int r = 0; r < 8; ++r)
    C[(size_t)(rbase + r) * N + colB] = (_Float16)(acc[r] + bv);
}

// ---------------------------------------------------------------- batchnorm
__global__ void bn_partial_kernel(const _Float16* __restrict__ X, int R, int C,
                                  float* __restrict__ sum,
                                  float* __restrict__ sumsq) {
  __shared__ float sh[256], sh2[256];
  int c = blockIdx.x;
  float s = 0.f, ss = 0.f;
  for (int r = blockIdx.y * blockDim.x + threadIdx.x; r < R;
       r += gridDim.y * blockDim.x) {
    float v = (float)X[(size_t)r * C + c];
    s += v; ss += v * v;
  }
  sh[threadIdx.x] = s; sh2[threadIdx.x] = ss;
  __syncthreads();
  for (int o = blockDim.x >> 1; o > 0; o >>= 1) {
    if (threadIdx.x < o) {
      sh[threadIdx.x]  += sh[threadIdx.x + o];
      sh2[threadIdx.x] += sh2[threadIdx.x + o];
    }
    __syncthreads();
  }
  if (threadIdx.x == 0) {
    atomicAdd(&sum[c], sh[0]);
    atomicAdd(&sumsq[c], sh2[0]);
  }
}

__global__ void bn_finalize_kernel(const float* __restrict__ sum,
                                   const float* __restrict__ sumsq,
                                   const float* __restrict__ g,
                                   const float* __restrict__ bt,
                                   int R, int C,
                                   float* __restrict__ scale,
                                   float* __restrict__ shift) {
  int c = blockIdx.x * blockDim.x + threadIdx.x;
  if (c >= C) return;
  float m = sum[c] / (float)R;
  float v = sumsq[c] / (float)R - m * m;
  float sc = g[c] * rsqrtf(v + 1e-5f);
  scale[c] = sc;
  shift[c] = bt[c] - m * sc;
}

__global__ void bn_relu_apply_kernel(_Float16* __restrict__ X, size_t n, int C,
                                     const float* __restrict__ scale,
                                     const float* __restrict__ shift) {
  size_t i = (size_t)blockIdx.x * blockDim.x + threadIdx.x;
  if (i >= n) return;
  int c = (int)(i % (size_t)C);
  float v = (float)X[i] * scale[c] + shift[c];
  X[i] = (_Float16)(v > 0.f ? v : 0.f);
}

// max over K samples; scatter into concat slot [coff, coff+C) of outC channels
__global__ void maxpool_k_kernel(const _Float16* __restrict__ X,
                                 _Float16* __restrict__ out,
                                 int outer, int K, int C, int outC, int coff) {
  int i = blockIdx.x * blockDim.x + threadIdx.x;
  if (i >= outer * C) return;
  int o = i / C, c = i - o * C;
  const _Float16* p = X + (size_t)o * K * C + c;
  float m = -3.4e38f;
  for (int k = 0; k < K; ++k) m = fmaxf(m, (float)p[(size_t)k * C]);
  out[(size_t)o * outC + coff + c] = (_Float16)m;
}

// final 256 -> 1 head
__global__ void fc3_kernel(const _Float16* __restrict__ h,
                           const float* __restrict__ W,
                           const float* __restrict__ b,
                           float* __restrict__ out, int B, int Cin) {
  int i = threadIdx.x;
  if (i >= B) return;
  float s = b[0];
  for (int c = 0; c < Cin; ++c) s += (float)h[(size_t)i * Cin + c] * W[c];
  out[i] = s;
}

// ================================================================== host ===
extern "C" void kernel_launch(void* const* d_in, const int* in_sizes, int n_in,
                              void* d_out, int out_size, void* d_ws,
                              size_t ws_size, hipStream_t stream) {
  (void)n_in; (void)out_size; (void)ws_size;
  const int B = 16, N0 = 4096;

  // ---- canonical layer table (24 layers) ----
  int cins[24], couts[24];
  {
    const int sa1m[3][3] = {{32, 32, 64}, {64, 64, 128}, {64, 96, 128}};
    const int sa2m[3][3] = {{64, 64, 128}, {128, 128, 256}, {128, 128, 256}};
    const int sa3m[3] = {256, 512, 1024};
    int li = 0;
    for (int br = 0; br < 3; ++br) {
      int cin = 3;
      for (int l = 0; l < 3; ++l) { cins[li] = cin; couts[li] = sa1m[br][l]; cin = couts[li]; ++li; }
    }
    for (int br = 0; br < 3; ++br) {
      int cin = 323;
      for (int l = 0; l < 3; ++l) { cins[li] = cin; couts[li] = sa2m[br][l]; cin = couts[li]; ++li; }
    }
    { int cin = 643; for (int l = 0; l < 3; ++l) { cins[li] = cin; couts[li] = sa3m[l]; cin = couts[li]; ++li; } }
    cins[21] = 1024; couts[21] = 512;
    cins[22] = 512;  couts[22] = 256;
    cins[23] = 256;  couts[23] = 1;
  }

  // ---- map flattened inputs (insertion order vs jax sorted-key order) ----
  const float *Wp[24], *bp[24], *gp[24], *btp[24];
  const float* xyz;
  bool insertionOrder = (in_sizes[0] == B * 3 * N0);
  if (insertionOrder) {
    xyz = (const float*)d_in[0];
    for (int i = 0; i < 23; ++i) {
      int base = 1 + 4 * i;
      Wp[i]  = (const float*)d_in[base + 0];
      bp[i]  = (const float*)d_in[base + 1];
      gp[i]  = (const float*)d_in[base + 2];
      btp[i] = (const float*)d_in[base + 3];
    }
    Wp[23] = (const float*)d_in[1 + 4 * 23];
    bp[23] = (const float*)d_in[1 + 4 * 23 + 1];
    gp[23] = btp[23] = nullptr;
  } else {
    auto set4 = [&](int li, int base) {
      Wp[li]  = (const float*)d_in[base + 0];
      bp[li]  = (const float*)d_in[base + 1];
      btp[li] = (const float*)d_in[base + 2];
      gp[li]  = (const float*)d_in[base + 3];
    };
    set4(21, 0); set4(22, 4);
    Wp[23] = (const float*)d_in[8];
    bp[23] = (const float*)d_in[9];
    gp[23] = btp[23] = nullptr;
    for (int j = 0; j < 9; ++j) set4(j, 10 + 4 * j);
    for (int j = 0; j < 9; ++j) set4(9 + j, 46 + 4 * j);
    for (int j = 0; j < 3; ++j) set4(18 + j, 82 + 4 * j);
    xyz = (const float*)d_in[94];
  }

  // ---- workspace bump allocator ----
  char* ws = (char*)d_ws;
  size_t off = 0;
  auto alloc = [&](size_t bytes) -> void* {
    off = (off + 255) & ~(size_t)255;
    void* p = ws + off;
    off += bytes;
    return p;
  };
  typedef _Float16 h16;

  float* pts    = (float*)alloc((size_t)B * N0 * 3 * sizeof(float));
  float* nx1    = (float*)alloc((size_t)B * 512 * 3 * sizeof(float));
  float* nx2    = (float*)alloc((size_t)B * 128 * 3 * sizeof(float));
  h16*   l1     = (h16*)alloc((size_t)B * 512 * 320 * sizeof(h16));
  h16*   l2     = (h16*)alloc((size_t)B * 128 * 640 * sizeof(h16));
  int*   fpsidx = (int*)alloc((size_t)B * 512 * sizeof(int));
  int*   ballix = (int*)alloc((size_t)B * 512 * 128 * sizeof(int));
  h16*   feat1k = (h16*)alloc((size_t)B * 1024 * sizeof(h16));
  h16*   h1     = (h16*)alloc((size_t)B * 512 * sizeof(h16));
  h16*   h2     = (h16*)alloc((size_t)B * 256 * sizeof(h16));
  h16*   wpack  = (h16*)alloc((size_t)1024 * 1024 * sizeof(h16));
  float* sumb   = (float*)alloc(1024 * sizeof(float));
  float* sqb    = (float*)alloc(1024 * sizeof(float));
  float* scaleb = (float*)alloc(1024 * sizeof(float));
  float* shiftb = (float*)alloc(1024 * sizeof(float));
  size_t bigAe = (size_t)1048576 * 96;                 // SA1-b2 96ch layer
  if ((size_t)262144 * 352 > bigAe) bigAe = (size_t)262144 * 352;
  size_t bigBe = (size_t)1048576 * 128;                // SA1-b2 128ch layer
  h16* bigA = (h16*)alloc(bigAe * sizeof(h16));
  h16* bigB = (h16*)alloc(bigBe * sizeof(h16));

  // ---- one conv-bn-relu layer (f16 in -> f16 out) ----
  auto runLayer = [&](const h16* in, h16* out, int rows, int li) {
    int cin = cins[li], cout = couts[li];
    int kpad = (cin + 31) & ~31;
    int numKc = kpad >> 5;
    int packTotal = (cout >> 4) * numKc * 512;
    pack_w_kernel<<<(packTotal + 255) / 256, 256, 0, stream>>>(Wp[li], wpack,
                                                               cin, cout, kpad);
    dim3 g(rows / 16, (cout + 63) / 64);
    wmma_gemm_bias_kernel<<<g, 128, 0, stream>>>(in, wpack, bp[li], out, rows,
                                                 kpad, cout);
    zero_kernel<<<(cout + 255) / 256, 256, 0, stream>>>(sumb, cout);
    zero_kernel<<<(cout + 255) / 256, 256, 0, stream>>>(sqb, cout);
    int splits = (rows + 255) / 256;
    if (splits > 64) splits = 64;
    if (splits < 1) splits = 1;
    bn_partial_kernel<<<dim3(cout, splits), 256, 0, stream>>>(out, rows, cout,
                                                              sumb, sqb);
    bn_finalize_kernel<<<(cout + 255) / 256, 256, 0, stream>>>(
        sumb, sqb, gp[li], btp[li], rows, cout, scaleb, shiftb);
    size_t n = (size_t)rows * cout;
    bn_relu_apply_kernel<<<(unsigned)((n + 255) / 256), 256, 0, stream>>>(
        out, n, cout, scaleb, shiftb);
  };

  // ---------------- stage 0: transpose ----------------
  transpose_xyz_kernel<<<(B * N0 + 255) / 256, 256, 0, stream>>>(xyz, pts, B, N0);

  // ---------------- SA1 (N=4096 -> M=512, MSG) ----------------
  fps_kernel<<<B, 256, 0, stream>>>(pts, N0, 512, fpsidx);
  gather_xyz_kernel<<<(B * 512 + 255) / 256, 256, 0, stream>>>(pts, fpsidx, nx1,
                                                               N0, 512, B);
  {
    const float radii[3] = {0.1f, 0.2f, 0.4f};
    const int Ks[3] = {16, 32, 128};
    const int coff[3] = {0, 64, 192};
    for (int br = 0; br < 3; ++br) {
      int K = Ks[br];
      float r2 = radii[br] * radii[br];
      ballquery_kernel<<<(B * 512 + 255) / 256, 256, 0, stream>>>(
          pts, nx1, ballix, B, N0, 512, K, r2);
      int rows = B * 512 * K;
      group_kernel<<<(rows + 255) / 256, 256, 0, stream>>>(
          pts, (const h16*)nullptr, nx1, ballix, bigA, B, N0, 512, K, 0, 32);
      h16 *cur = bigA, *nxt = bigB;
      for (int l = 0; l < 3; ++l) {
        runLayer(cur, nxt, rows, br * 3 + l);
        h16* t = cur; cur = nxt; nxt = t;
      }
      int Cl = couts[br * 3 + 2];
      maxpool_k_kernel<<<(B * 512 * Cl + 255) / 256, 256, 0, stream>>>(
          cur, l1, B * 512, K, Cl, 320, coff[br]);
    }
  }

  // ---------------- SA2 (N=512 -> M=128, MSG, feat=320) ----------------
  fps_kernel<<<B, 256, 0, stream>>>(nx1, 512, 128, fpsidx);
  gather_xyz_kernel<<<(B * 128 + 255) / 256, 256, 0, stream>>>(nx1, fpsidx, nx2,
                                                               512, 128, B);
  {
    const float radii[3] = {0.2f, 0.4f, 0.8f};
    const int Ks[3] = {32, 64, 128};
    const int coff[3] = {0, 128, 384};
    for (int br = 0; br < 3; ++br) {
      int K = Ks[br];
      float r2 = radii[br] * radii[br];
      ballquery_kernel<<<(B * 128 + 255) / 256, 256, 0, stream>>>(
          nx1, nx2, ballix, B, 512, 128, K, r2);
      int rows = B * 128 * K;
      group_kernel<<<(rows + 255) / 256, 256, 0, stream>>>(
          nx1, l1, nx2, ballix, bigA, B, 512, 128, K, 320, 352);
      h16 *cur = bigA, *nxt = bigB;
      for (int l = 0; l < 3; ++l) {
        runLayer(cur, nxt, rows, 9 + br * 3 + l);
        h16* t = cur; cur = nxt; nxt = t;
      }
      int Cl = couts[9 + br * 3 + 2];
      maxpool_k_kernel<<<(B * 128 * Cl + 255) / 256, 256, 0, stream>>>(
          cur, l2, B * 128, K, Cl, 640, coff[br]);
    }
  }

  // ---------------- SA3 (group-all over 128 pts, 643ch -> pad 672) ----------
  {
    int rows = B * 128;  // 2048
    concat_sa3_kernel<<<(rows + 255) / 256, 256, 0, stream>>>(nx2, l2, bigA, rows);
    h16 *cur = bigA, *nxt = bigB;
    for (int l = 0; l < 3; ++l) {
      runLayer(cur, nxt, rows, 18 + l);
      h16* t = cur; cur = nxt; nxt = t;
    }
    maxpool_k_kernel<<<(B * 1024 + 255) / 256, 256, 0, stream>>>(
        cur, feat1k, B, 128, 1024, 1024, 0);
  }

  // ---------------- FC head ----------------
  runLayer(feat1k, h1, B, 21);   // 1024 -> 512
  runLayer(h1, h2, B, 22);       // 512 -> 256
  fc3_kernel<<<1, 32, 0, stream>>>(h2, Wp[23], bp[23], (float*)d_out, B, 256);
}
